// TorchtutNet_70549132804660
// MI455X (gfx1250) — matrix operand order repaired and checked by reading, not compiled, not run on hardware
//
#include <hip/hip_runtime.h>

// CDNA5 / gfx1250: wave32, WMMA 16x16x32 f16 -> f32 accum.
typedef __attribute__((ext_vector_type(16))) _Float16 v16h;
typedef __attribute__((ext_vector_type(8)))  float    v8f;

#define KB 1024ull
#define MB (1024ull*1024ull)

// ---------------- workspace layout (bytes), ~101 MiB total ----------------
// Binarized, zero-padded f16 weight matrices (A operands, row-major [Mpad][Kpad]):
static const size_t WQ1_OFF  = 0;        // 16 x 96   (w1: 6x75)
static const size_t WQ2_OFF  = 8*KB;     // 16 x 160  (w2: 16x150)
static const size_t WF1_OFF  = 32*KB;    // 128 x 416 (fw1: 120x400)
static const size_t WF2_OFF  = 160*KB;   // 96 x 128  (fw2: 84x120)
static const size_t WF3_OFF  = 192*KB;   // 16 x 96   (fw3: 10x84)
// f16 activations (L2-resident on MI455X's 192MB L2). Lifetime-based reuse.
// All padded with genuine zeros so WMMA B-gathers need no per-element guards.
static const size_t ACT1_OFF = 1*MB;     // [8192][6][28][28]    73.5 MiB (dead after pool1)
static const size_t P1_OFF   = 76*MB;    // [8192][8][14][14]    24.5 MiB (c=6,7 zero)
static const size_t ACT2_OFF = 1*MB;     // [8192][16][10][10]   25 MiB (reuses act1 region)
static const size_t P2_OFF   = 30*MB;    // [8192][416]          6.5 MiB (k>=400 zero)
static const size_t H1_OFF   = 40*MB;    // [8192][128]          2 MiB (m>=120 zero)
static const size_t H2_OFF   = 44*MB;    // [8192][96]           1.5 MiB (m>=84 zero)

// A-fragment K map: contiguous order k = 16*h + e (matches B gathers).
__device__ inline int kmapA(int e, int h) { return 16 * h + e; }

__device__ inline _Float16 hmax4(_Float16 a, _Float16 b, _Float16 c, _Float16 d) {
  _Float16 m0 = (a > b) ? a : b;
  _Float16 m1 = (c > d) ? c : d;
  return (m0 > m1) ? m0 : m1;
}

// CDNA5 async global->LDS copy (ASYNCcnt-tracked); generic LDS ptr truncated to
// 32 bits == LDS byte offset.
__device__ inline void async_ld_b128(void* lds, const void* gaddr) {
  unsigned loff = (unsigned)(unsigned long long)lds;
  asm volatile("global_load_async_to_lds_b128 %0, %1, off"
               :: "v"(loff), "v"(gaddr) : "memory");
}
__device__ inline void wait_async0() {
  asm volatile("s_wait_asynccnt 0x0" ::: "memory");
}

// ------------- weight prep: scales + binarize + pad to f16 -------------
__device__ inline void writePad(char* dstp, const float* src, int Mr, int Kr,
                                int Mp, int Kp, float s, int tid) {
  _Float16* dst = (_Float16*)dstp;
  for (int idx = tid; idx < Mp * Kp; idx += 256) {
    int m = idx / Kp, k = idx % Kp;
    float v = 0.f;
    if (m < Mr && k < Kr) v = (src[m * Kr + k] >= 0.f) ? s : -s;
    dst[idx] = (_Float16)v;
  }
}

__global__ __launch_bounds__(256)
void prep_weights(const float* w1, const float* w2, const float* fw1,
                  const float* fw2, const float* fw3, char* ws) {
  __shared__ float red[256];
  __shared__ float sc[5];
  const int tid = threadIdx.x;
  const float* srcs[5] = {w1, w2, fw1, fw2, fw3};
  const int sizes[5]   = {450, 2400, 48000, 10080, 840};
  #pragma unroll
  for (int t = 0; t < 5; ++t) {
    float s = 0.f;
    for (int i = tid; i < sizes[t]; i += 256) s += fabsf(srcs[t][i]);
    red[tid] = s;
    __syncthreads();
    for (int off = 128; off > 0; off >>= 1) {
      if (tid < off) red[tid] += red[tid + off];
      __syncthreads();
    }
    if (tid == 0) sc[t] = red[0] / (float)sizes[t];
    __syncthreads();
  }
  writePad(ws + WQ1_OFF, w1, 6, 75, 16, 96, sc[0], tid);
  writePad(ws + WQ2_OFF, w2, 16, 150, 16, 160, sc[1], tid);
  writePad(ws + WF1_OFF, fw1, 120, 400, 128, 416, sc[2], tid);
  writePad(ws + WF2_OFF, fw2, 84, 120, 96, 128, sc[3], tid);
  writePad(ws + WF3_OFF, fw3, 10, 84, 16, 96, sc[4], tid);
}

// ------------- conv1: LDS-tiled WMMA, one workgroup per image -------------
// Stage 1: async-copy x[b] (3x32x32 f32) into LDS (ASYNCcnt path).
// Stage 2: convert own chunks to a f16 LDS image with zeroed 4th channel.
// Compute: B-gather base offsets are half-selected compile-time constants,
// precomputed once per wave -> inner loop is one v_add + one ds_load_u16 each.
__global__ __launch_bounds__(256)
void conv1_wmma(const float* __restrict__ x, const float* __restrict__ bias, char* ws) {
  __shared__ __align__(16) float    xs[3072];   // staged f32 [3][32][32]
  __shared__ __align__(16) _Float16 xsh[4096];  // f16 [4][32][32], ch3 = 0
  const int b   = blockIdx.x;
  const int tid = threadIdx.x;
  const float* xg = x + (size_t)b * 3072;
  #pragma unroll
  for (int i = 0; i < 3; ++i) {
    int e4 = tid + i * 256;                     // float4 chunk 0..767
    async_ld_b128(&xs[e4 * 4], xg + e4 * 4);
  }
  #pragma unroll
  for (int i = 0; i < 4; ++i)                   // zero f16 channel 3
    xsh[3072 + tid + i * 256] = (_Float16)0.f;
  wait_async0();                                // own wave's async copies done
  #pragma unroll
  for (int i = 0; i < 3; ++i) {                 // convert own chunks f32 -> f16
    int base = (tid + i * 256) * 4;
    float4 v4 = *(const float4*)&xs[base];
    xsh[base + 0] = (_Float16)v4.x;
    xsh[base + 1] = (_Float16)v4.y;
    xsh[base + 2] = (_Float16)v4.z;
    xsh[base + 3] = (_Float16)v4.w;
  }
  __syncthreads();

  const int lane = tid & 31;
  const int wave = tid >> 5;
  const int half = lane >> 4;
  const int col  = lane & 15;

  // Preload the 3 A fragments (weights, K=96) once per wave.
  const _Float16* A = (const _Float16*)(ws + WQ1_OFF);
  v16h a[3];
  #pragma unroll
  for (int kb = 0; kb < 3; ++kb)
    #pragma unroll
    for (int e = 0; e < 16; ++e)
      a[kb][e] = A[(lane & 15) * 96 + kb * 32 + kmapA(e, half)];

  // Half-selected k-derived LDS element offsets (constants after unroll).
  int iofs[3][16];
  #pragma unroll
  for (int kb = 0; kb < 3; ++kb)
    #pragma unroll
    for (int e = 0; e < 16; ++e) {
      int k0 = kb * 32 + e, k1 = k0 + 16;       // k for half 0 / half 1
      int r0 = k0 % 25,     r1 = k1 % 25;
      int c0 = (k0 / 25) * 1024 + (r0 / 5) * 32 + (r0 % 5);
      int c1 = (k1 / 25) * 1024 + (r1 / 5) * 32 + (r1 % 5);
      iofs[kb][e] = half ? c1 : c0;
    }

  _Float16* act1 = (_Float16*)(ws + ACT1_OFF);
  for (int t = wave; t < 49; t += 8) {          // 49 N-tiles of 16 per image
    const int nloc = t * 16 + col;
    const int oh = nloc / 28, ow = nloc % 28;
    const int pix = oh * 32 + ow;
    v8f acc = {};
    #pragma unroll
    for (int kb = 0; kb < 3; ++kb) {
      v16h bv;
      #pragma unroll
      for (int e = 0; e < 16; ++e)
        bv[e] = xsh[pix + iofs[kb][e]];         // v_add + ds_load_u16
      acc = __builtin_amdgcn_wmma_f32_16x16x32_f16(false, a[kb], false, bv,
                                                   (short)0, acc, false, false);
    }
    // C/D layout: M = r + 8*half. Only M<6 exists -> only half==0 stores, and
    // bias[r] is wave-uniform (scalar load). Single predicated region.
    if (half == 0) {
      #pragma unroll
      for (int r = 0; r < 6; ++r) {
        float v = fmaxf(acc[r] + bias[r], 0.f);
        act1[((size_t)(b * 6 + r) * 28 + oh) * 28 + ow] = (_Float16)v;
      }
    }
  }
}

// ------------- pool1: act1 -> channel-padded [8192][8][14][14] -------------
__global__ __launch_bounds__(256)
void pool1_k(const _Float16* __restrict__ in, _Float16* __restrict__ out) {
  int idx = blockIdx.x * 256 + threadIdx.x;  // over 8192*8*196
  int j  = idx % 196;
  int c  = (idx / 196) % 8;
  int b  = idx / (8 * 196);
  _Float16 v = (_Float16)0.f;
  if (c < 6) {
    int oh = j / 14, ow = j % 14;
    const _Float16* p = in + (((size_t)(b * 6 + c) * 28 + 2 * oh) * 28 + 2 * ow);
    v = hmax4(p[0], p[1], p[28], p[29]);
  }
  out[idx] = v;
}

// ------------- pool2: act2 -> K-padded flat [8192][416] -------------
__global__ __launch_bounds__(256)
void pool2_k(const _Float16* __restrict__ in, _Float16* __restrict__ out) {
  int idx = blockIdx.x * 256 + threadIdx.x;  // over 8192*416
  int j = idx % 416;
  int n = idx / 416;
  _Float16 v = (_Float16)0.f;
  if (j < 400) {
    int c = j / 25, r = (j % 25) / 5, q = j % 5;
    const _Float16* p = in + (((size_t)(n * 16 + c) * 10 + 2 * r) * 10 + 2 * q);
    v = hmax4(p[0], p[1], p[10], p[11]);
  }
  out[idx] = v;
}

// ------------- layers 2-5: generic GEMM on v_wmma_f32_16x16x32_f16 -------------
// All B sources are zero-padded, so gathers are branch-free for the full KPAD.
template <int L>
__device__ inline _Float16 gatherB(int k, int n, const char* ws) {
  if constexpr (L == 3) {               // fc1: pool2 flat [8192][416], contiguous K
    return ((const _Float16*)(ws + P2_OFF))[n * 416 + k];
  } else if constexpr (L == 4) {        // fc2: h1 [8192][128]
    return ((const _Float16*)(ws + H1_OFF))[n * 128 + k];
  } else {                              // fc3: h2 [8192][96]
    return ((const _Float16*)(ws + H2_OFF))[n * 96 + k];
  }
}

template <int L>
__global__ __launch_bounds__(256)
void gemm_wmma(const float* __restrict__ bias, char* ws, float* out) {
  constexpr int MREAL  = (L == 2) ? 16 : (L == 3) ? 120 : (L == 4) ? 84 : 10;
  constexpr int MTILES = (L == 3) ? 8 : (L == 4) ? 6 : 1;
  constexpr int KPAD   = (L == 2) ? 160 : (L == 3) ? 416 : (L == 4) ? 128 : 96;
  constexpr int NKB    = KPAD / 32;
  constexpr size_t AOFF = (L == 2) ? WQ2_OFF : (L == 3) ? WF1_OFF
                        : (L == 4) ? WF2_OFF : WF3_OFF;

  const int lane  = threadIdx.x & 31;
  const int wave  = threadIdx.x >> 5;
  const int tile  = blockIdx.x * (blockDim.x >> 5) + wave;
  const int mtile = tile % MTILES;
  const int ntile = tile / MTILES;
  const int half  = lane >> 4;
  const int n     = ntile * 16 + (lane & 15);
  const int arow  = mtile * 16 + (lane & 15);

  // conv2 im2col: half-selected k-derived offsets into padded pool1 image
  // (constants after unroll) + per-lane image base computed once.
  int iofs2[NKB][16];
  int imgbase = 0;
  if constexpr (L == 2) {
    int nb = n / 100, s = n % 100;
    imgbase = nb * 1568 + (s / 10) * 14 + (s % 10);   // [8192][8][14][14]
    #pragma unroll
    for (int kb = 0; kb < NKB; ++kb)
      #pragma unroll
      for (int e = 0; e < 16; ++e) {
        int k0 = kb * 32 + e, k1 = k0 + 16;
        int r0 = k0 % 25,     r1 = k1 % 25;
        int c0 = (k0 / 25) * 196 + (r0 / 5) * 14 + (r0 % 5);
        int c1 = (k1 / 25) * 196 + (r1 / 5) * 14 + (r1 % 5);
        iofs2[kb][e] = half ? c1 : c0;
      }
  }
  const _Float16* p1 = (const _Float16*)(ws + P1_OFF);

  const _Float16* A = (const _Float16*)(ws + AOFF);
  v8f acc = {};
  #pragma unroll
  for (int kb = 0; kb < NKB; ++kb) {
    v16h av, bv;
    #pragma unroll
    for (int e = 0; e < 16; ++e) {
      av[e] = A[arow * KPAD + kb * 32 + kmapA(e, half)];
      if constexpr (L == 2)
        bv[e] = p1[imgbase + iofs2[kb][e]];           // v_add + global u16 load
      else
        bv[e] = gatherB<L>(kb * 32 + 16 * half + e, n, ws);
    }
    acc = __builtin_amdgcn_wmma_f32_16x16x32_f16(false, av, false, bv,
                                                 (short)0, acc, false, false);
  }
  // Epilogue. C/D layout: M = mtile*16 + r + 8*half, N = lane&15.
  // Layers 2-4 store branchlessly into zero-padded buffers (bias index clamped
  // so the load stays in-bounds; padded rows get true zeros). fc3 guards stores.
  #pragma unroll
  for (int r = 0; r < 8; ++r) {
    int m = mtile * 16 + r + 8 * half;
    if constexpr (L == 5) {
      if ((r < 2) || (half == 0)) {          // m<10 exactly (r<2 is compile-time)
        out[n * 10 + m] = acc[r] + bias[m];
      }
    } else {
      int mi = (m < MREAL) ? m : 0;          // clamp for in-bounds bias read
      float v = fmaxf(acc[r] + bias[mi], 0.f);
      if (m >= MREAL) v = 0.f;               // true zeros into padding
      if constexpr (L == 2) {
        int b = n / 100, s = n % 100, oh = s / 10, ow = s % 10;
        ((_Float16*)(ws + ACT2_OFF))[((b * 16 + m) * 10 + oh) * 10 + ow] = (_Float16)v;
      } else if constexpr (L == 3) {
        ((_Float16*)(ws + H1_OFF))[n * 128 + m] = (_Float16)v;
      } else {
        ((_Float16*)(ws + H2_OFF))[n * 96 + m] = (_Float16)v;
      }
    }
  }
}

// ------------------------------- launch -------------------------------
extern "C" void kernel_launch(void* const* d_in, const int* in_sizes, int n_in,
                              void* d_out, int out_size, void* d_ws, size_t ws_size,
                              hipStream_t stream) {
  (void)in_sizes; (void)n_in; (void)out_size; (void)ws_size;
  const float* x   = (const float*)d_in[0];
  const float* w1  = (const float*)d_in[1];
  const float* b1  = (const float*)d_in[2];
  const float* w2  = (const float*)d_in[3];
  const float* b2  = (const float*)d_in[4];
  const float* fw1 = (const float*)d_in[5];
  const float* fb1 = (const float*)d_in[6];
  const float* fw2 = (const float*)d_in[7];
  const float* fb2 = (const float*)d_in[8];
  const float* fw3 = (const float*)d_in[9];
  const float* fb3 = (const float*)d_in[10];
  char* ws   = (char*)d_ws;
  float* out = (float*)d_out;

  prep_weights<<<1, 256, 0, stream>>>(w1, w2, fw1, fw2, fw3, ws);

  conv1_wmma<<<8192, 256, 0, stream>>>(x, b1, ws);                  // -> act1

  pool1_k<<<50176, 256, 0, stream>>>((const _Float16*)(ws + ACT1_OFF),
                                     (_Float16*)(ws + P1_OFF));     // 8192*8*196

  gemm_wmma<2><<<6400, 256, 0, stream>>>(b2, ws, out);              // N=819200 -> act2

  pool2_k<<<13312, 256, 0, stream>>>((const _Float16*)(ws + ACT2_OFF),
                                     (_Float16*)(ws + P2_OFF));     // 8192*416

  gemm_wmma<3><<<512, 256, 0, stream>>>(fb1, ws, out);              // 8 x 512 tiles -> h1
  gemm_wmma<4><<<384, 256, 0, stream>>>(fb2, ws, out);              // 6 x 512 tiles -> h2
  gemm_wmma<5><<<64,  256, 0, stream>>>(fb3, ws, out);              // 1 x 512 tiles -> out
}
